// ParallelSTBlock_35278861369726
// MI455X (gfx1250) — matrix operand order repaired and checked by reading, not compiled, not run on hardware
//
#include <hip/hip_runtime.h>
#include <hip/hip_bf16.h>

typedef _Float16 half_t;
typedef __attribute__((ext_vector_type(16))) _Float16 v16h;
typedef __attribute__((ext_vector_type(8)))  _Float16 v8h;
typedef __attribute__((ext_vector_type(8)))  float    v8f;

// ---- problem constants (match reference) ----
constexpr int B_  = 8;
constexpr int N_  = 200;
constexpr int T_  = 12;
constexpr int H_  = 128;
constexpr int HEADS_ = 4;
constexpr int E_  = 3200;
constexpr int ETOT_ = E_ + N_;     // 3400 (edges + self loops)
constexpr int BT_ = B_ * T_;       // 96
constexpr int M_  = BT_ * N_;      // 19200 rows for all GEMMs (== B*N*T)

// ============================================================
// f16 WMMA GEMM with 2x2 register blocking:
//   out[M,N] = A[M,K] * W[N,K]^T + bias
// One wave per 32x32 output tile (4 accumulators). Each A fragment
// is reused across 2 N-tiles and each B fragment across 2 M-tiles:
// 8 b128 loads feed 4 v_wmma per k-step (vs 4 loads per wmma for a
// 16x16 tile). Operands are loaded straight from global (entire
// working set is L2-resident on MI455X's 192MB L2).
// ACT: 0 = none, 1 = exact GELU. Optional fused f16 store.
// ============================================================
template<int ACT>
__global__ __launch_bounds__(256) void k_gemm_wmma(
    const half_t* __restrict__ A, const half_t* __restrict__ W,
    const float* __restrict__ bias,
    float* __restrict__ outF, half_t* __restrict__ outH,
    int M, int N, int K)
{
  int wave = blockIdx.x * (blockDim.x >> 5) + (threadIdx.x >> 5);
  int tilesN = N >> 5;
  int numTiles = (M >> 5) * tilesN;
  if (wave >= numTiles) return;           // wave-uniform, EXEC stays all-ones
  int tm = wave / tilesN;
  int tn = wave % tilesN;

  int lane = threadIdx.x & 31;
  int sub  = lane >> 4;                   // lane half
  int r    = lane & 15;

  // A fragment rows: per lane-half k-chunks [sub*8, +8) and [16+sub*8, +8)
  const half_t* Arow0 = A + (size_t)(tm * 32 + r) * K + sub * 8;
  const half_t* Arow1 = Arow0 + (size_t)16 * K;
  // B = W^T: lane-half covers column n, k-chunks [sub*8, +8) and [16+sub*8, +8)
  const half_t* Brow0 = W + (size_t)(tn * 32 + r) * K + sub * 8;
  const half_t* Brow1 = Brow0 + (size_t)16 * K;

  float bv0 = bias ? bias[tn * 32 + r]      : 0.0f;
  float bv1 = bias ? bias[tn * 32 + 16 + r] : 0.0f;
  v8f acc00, acc01, acc10, acc11;
#pragma unroll
  for (int i = 0; i < 8; ++i) {
    acc00[i] = bv0; acc10[i] = bv0;
    acc01[i] = bv1; acc11[i] = bv1;
  }

  for (int kb = 0; kb < K; kb += 32) {
    v8h a00 = *(const v8h*)(Arow0 + kb);
    v8h a01 = *(const v8h*)(Arow0 + kb + 16);
    v8h a10 = *(const v8h*)(Arow1 + kb);
    v8h a11 = *(const v8h*)(Arow1 + kb + 16);
    v8h b00 = *(const v8h*)(Brow0 + kb);
    v8h b01 = *(const v8h*)(Brow0 + kb + 16);
    v8h b10 = *(const v8h*)(Brow1 + kb);
    v8h b11 = *(const v8h*)(Brow1 + kb + 16);
    v16h va0, va1, vb0, vb1;
#pragma unroll
    for (int i = 0; i < 8; ++i) {
      va0[i] = a00[i]; va0[i + 8] = a01[i];
      va1[i] = a10[i]; va1[i + 8] = a11[i];
      vb0[i] = b00[i]; vb0[i + 8] = b01[i];
      vb1[i] = b10[i]; vb1[i + 8] = b11[i];
    }
    acc00 = __builtin_amdgcn_wmma_f32_16x16x32_f16(false, va0, false, vb0, (short)0, acc00, false, false);
    acc01 = __builtin_amdgcn_wmma_f32_16x16x32_f16(false, va0, false, vb1, (short)0, acc01, false, false);
    acc10 = __builtin_amdgcn_wmma_f32_16x16x32_f16(false, va1, false, vb0, (short)0, acc10, false, false);
    acc11 = __builtin_amdgcn_wmma_f32_16x16x32_f16(false, va1, false, vb1, (short)0, acc11, false, false);
  }

  auto epi = [&](v8f acc, int moff, int noff) {
    int n = tn * 32 + noff + r;
#pragma unroll
    for (int i = 0; i < 8; ++i) {
      int m = tm * 32 + moff + i + 8 * sub;  // C/D layout: VGPR i -> M=i (lanes<16) / i+8
      float v = acc[i];
      if (ACT == 1) v = 0.5f * v * (1.0f + erff(v * 0.70710678118654752f));
      size_t off = (size_t)m * N + n;
      outF[off] = v;
      if (outH) outH[off] = (half_t)v;
    }
  };
  epi(acc00,  0,  0);
  epi(acc01,  0, 16);
  epi(acc10, 16,  0);
  epi(acc11, 16, 16);
}

// ============================================================
// Small utility kernels
// ============================================================
__global__ void k_zero(float* __restrict__ p, long long count) {
  long long i = (long long)blockIdx.x * blockDim.x + threadIdx.x;
  for (; i < count; i += (long long)gridDim.x * blockDim.x) p[i] = 0.0f;
}

__global__ void k_f32_to_f16(const float* __restrict__ s, half_t* __restrict__ d, long long count) {
  long long i = (long long)blockIdx.x * blockDim.x + threadIdx.x;
  for (; i < count; i += (long long)gridDim.x * blockDim.x) d[i] = (half_t)s[i];
}

// xn16 = f16(x) flat; xt16[bt*N+n, h] = f16(x[b,n,t,h]) with bt = b*T+t
__global__ void k_build_inputs(const float* __restrict__ x,
                               half_t* __restrict__ xt16, half_t* __restrict__ xn16) {
  int idx = blockIdx.x * 256 + threadIdx.x;
  if (idx >= M_ * H_) return;
  xn16[idx] = (half_t)x[idx];
  int h = idx & (H_ - 1);
  int m = idx >> 7;
  int n  = m % N_;
  int bt = m / N_;
  int b  = bt / T_;
  int t  = bt % T_;
  size_t s = (((size_t)(b * N_ + n)) * T_ + t) * H_ + h;
  xt16[idx] = (half_t)x[s];
}

// ============================================================
// GAT edge kernels (segment softmax via atomics; max-sub omitted
// which is mathematically identical for bounded scores)
// ============================================================
__global__ __launch_bounds__(256) void k_edge_score(
    const float* __restrict__ xl, const float* __restrict__ xr,
    const int* __restrict__ ei, const float* __restrict__ att,
    float* __restrict__ wbuf, float* __restrict__ den)
{
  int unit = blockIdx.x * 8 + (threadIdx.x >> 5);   // (edge, bt)
  if (unit >= ETOT_ * BT_) return;
  int e  = unit / BT_;
  int bt = unit % BT_;
  int lane = threadIdx.x & 31;
  int src, dst; bool valid;
  if (e < E_) { src = ei[e]; dst = ei[E_ + e]; valid = (src != dst); }
  else        { src = dst = e - E_; valid = true; }

  const float* xls = xl + ((size_t)bt * N_ + src) * (HEADS_ * H_);
  const float* xrd = xr + ((size_t)bt * N_ + dst) * (HEADS_ * H_);
#pragma unroll
  for (int h = 0; h < HEADS_; ++h) {
    float s = 0.0f;
#pragma unroll
    for (int i = 0; i < 4; ++i) {
      int t = lane + 32 * i;
      float v = xls[h * H_ + t] + xrd[h * H_ + t];
      v = (v >= 0.0f) ? v : 0.2f * v;            // leaky_relu 0.2
      s += v * att[h * H_ + t];
    }
    for (int off = 16; off > 0; off >>= 1) s += __shfl_xor(s, off, 32);
    if (lane == 0) {
      float w = valid ? __expf(s) : 0.0f;
      wbuf[((size_t)e * BT_ + bt) * HEADS_ + h] = w;
      atomicAdd(&den[((size_t)dst * BT_ + bt) * HEADS_ + h], w);
    }
  }
}

__global__ __launch_bounds__(128) void k_edge_aggr(
    const float* __restrict__ xl, const float* __restrict__ wbuf,
    const float* __restrict__ den, const int* __restrict__ ei,
    float* __restrict__ acc)
{
  int unit = blockIdx.x;                   // (edge, bt)
  int e  = unit / BT_;
  int bt = unit % BT_;
  int t  = threadIdx.x;
  int src, dst;
  if (e < E_) { src = ei[e]; dst = ei[E_ + e]; if (src == dst) return; }
  else        { src = dst = e - E_; }
  const float* xls = xl + ((size_t)bt * N_ + src) * (HEADS_ * H_);
  float s = 0.0f;
#pragma unroll
  for (int h = 0; h < HEADS_; ++h) {
    float w  = wbuf[((size_t)e * BT_ + bt) * HEADS_ + h];
    float dn = den[((size_t)dst * BT_ + bt) * HEADS_ + h];
    s += (w / dn) * 0.25f * xls[h * H_ + t];   // 0.25 = mean over heads
  }
  atomicAdd(&acc[((size_t)dst * BT_ + bt) * H_ + t], s);
}

// ============================================================
// LayerNorm helpers (one 128-wide block per row)
// ============================================================
__device__ __forceinline__ float block_sum_128(float v, float* sbuf) {
  int tid = threadIdx.x;
  sbuf[tid] = v; __syncthreads();
  for (int s = 64; s > 0; s >>= 1) {
    if (tid < s) sbuf[tid] += sbuf[tid + s];
    __syncthreads();
  }
  float r = sbuf[0]; __syncthreads();
  return r;
}

__global__ __launch_bounds__(128) void k_add_ln(
    const float* __restrict__ a, const float* __restrict__ b,
    const float* __restrict__ gamma, const float* __restrict__ beta,
    float* __restrict__ outF, half_t* __restrict__ outH)
{
  __shared__ float sbuf[128];
  int m = blockIdx.x, h = threadIdx.x;
  size_t off = (size_t)m * H_ + h;
  float v = a[off] + (b ? b[off] : 0.0f);
  float mu  = block_sum_128(v, sbuf) * (1.0f / H_);
  float d   = v - mu;
  float var = block_sum_128(d * d, sbuf) * (1.0f / H_);
  float y = d * rsqrtf(var + 1e-5f) * gamma[h] + beta[h];
  outF[off] = y;
  if (outH) outH[off] = (half_t)y;
}

// x_spatial[b,n,t,:] = LN(acc[n, b*T+t, :] + gat_bias)
__global__ __launch_bounds__(128) void k_spatial_ln(
    const float* __restrict__ acc, const float* __restrict__ gbias,
    const float* __restrict__ gamma, const float* __restrict__ beta,
    float* __restrict__ outF)
{
  __shared__ float sbuf[128];
  int m = blockIdx.x, h = threadIdx.x;
  int b = m / (N_ * T_);
  int rem = m % (N_ * T_);
  int n = rem / T_;
  int t = rem % T_;
  int bt = b * T_ + t;
  float v = acc[((size_t)n * BT_ + bt) * H_ + h] + gbias[h];
  float mu  = block_sum_128(v, sbuf) * (1.0f / H_);
  float d   = v - mu;
  float var = block_sum_128(d * d, sbuf) * (1.0f / H_);
  outF[(size_t)m * H_ + h] = d * rsqrtf(var + 1e-5f) * gamma[h] + beta[h];
}

// ============================================================
// Tiny temporal attention: T=12, DH=32, one block per (bn, head)
// ============================================================
__global__ __launch_bounds__(384) void k_attn(
    const float* __restrict__ qkv, float* __restrict__ o)
{
  __shared__ float ql[12][32], kl[12][32], vl[12][33], sc[12][12];
  int unit = blockIdx.x;
  int bn = unit >> 2, head = unit & 3;
  int tid = threadIdx.x;
  int q = tid >> 5;          // 0..11
  int d = tid & 31;          // 0..31
  size_t base = ((size_t)bn * T_ + q) * (3 * H_) + head * 32 + d;
  ql[q][d] = qkv[base];
  kl[q][d] = qkv[base + H_];
  vl[q][d] = qkv[base + 2 * H_];
  __syncthreads();
  if (d < 12) {
    float s = 0.0f;
#pragma unroll
    for (int i = 0; i < 32; ++i) s += ql[q][i] * kl[d][i];
    sc[q][d] = s * 0.17677669529663687f;   // 1/sqrt(32)
  }
  __syncthreads();
  float mx = -1e30f;
#pragma unroll
  for (int k2 = 0; k2 < 12; ++k2) mx = fmaxf(mx, sc[q][k2]);
  float p[12]; float sum = 0.0f;
#pragma unroll
  for (int k2 = 0; k2 < 12; ++k2) { p[k2] = __expf(sc[q][k2] - mx); sum += p[k2]; }
  float inv = 1.0f / sum, accv = 0.0f;
#pragma unroll
  for (int k2 = 0; k2 < 12; ++k2) accv += p[k2] * inv * vl[k2][d];
  o[((size_t)bn * T_ + q) * H_ + head * 32 + d] = accv;
}

// cat16[m, 0:128] = f16(x_sp), cat16[m, 128:256] = f16(x_tmp)
__global__ void k_concat16(const float* __restrict__ xsp, const float* __restrict__ xtmp,
                           half_t* __restrict__ cat) {
  long long i = (long long)blockIdx.x * 256 + threadIdx.x;
  if (i >= (long long)M_ * 256) return;
  int c = (int)(i & 255);
  long long m = i >> 8;
  float v = (c < H_) ? xsp[m * H_ + c] : xtmp[m * H_ + (c - H_)];
  cat[i] = (half_t)v;
}

// y = LN(sigmoid(apre)*x_sp + (1-sigmoid)*x_tmp + x)
__global__ __launch_bounds__(128) void k_gate_final(
    const float* __restrict__ apre, const float* __restrict__ xsp,
    const float* __restrict__ xtmp, const float* __restrict__ x,
    const float* __restrict__ gamma, const float* __restrict__ beta,
    float* __restrict__ out)
{
  __shared__ float sbuf[128];
  int m = blockIdx.x, h = threadIdx.x;
  size_t off = (size_t)m * H_ + h;
  float a = 1.0f / (1.0f + __expf(-apre[off]));
  float v = a * xsp[off] + (1.0f - a) * xtmp[off] + x[off];
  float mu  = block_sum_128(v, sbuf) * (1.0f / H_);
  float d   = v - mu;
  float var = block_sum_128(d * d, sbuf) * (1.0f / H_);
  out[off] = d * rsqrtf(var + 1e-5f) * gamma[h] + beta[h];
}

// ============================================================
// Host launcher
// ============================================================
static inline int gemm_grid(int M, int N) {
  int tiles = (M / 32) * (N / 32);
  return (tiles + 7) / 8;                  // 8 waves / 256-thread block
}

extern "C" void kernel_launch(void* const* d_in, const int* in_sizes, int n_in,
                              void* d_out, int out_size, void* d_ws, size_t ws_size,
                              hipStream_t stream) {
  (void)in_sizes; (void)n_in; (void)out_size; (void)ws_size;

  const float* x        = (const float*)d_in[0];
  const int*   ei       = (const int*)  d_in[1];
  const float* gat_wl   = (const float*)d_in[2];
  const float* gat_bl   = (const float*)d_in[3];
  const float* gat_wr   = (const float*)d_in[4];
  const float* gat_br   = (const float*)d_in[5];
  const float* gat_att  = (const float*)d_in[6];
  const float* gat_bias = (const float*)d_in[7];
  const float* ain_w    = (const float*)d_in[8];
  const float* ain_b    = (const float*)d_in[9];
  const float* aout_w   = (const float*)d_in[10];
  const float* aout_b   = (const float*)d_in[11];
  const float* w1       = (const float*)d_in[12];
  const float* b1       = (const float*)d_in[13];
  const float* w2       = (const float*)d_in[14];
  const float* b2       = (const float*)d_in[15];
  const float* gw       = (const float*)d_in[16];
  const float* gb       = (const float*)d_in[17];
  const float* ln_sp_w  = (const float*)d_in[18];
  const float* ln_sp_b  = (const float*)d_in[19];
  const float* ln_t1_w  = (const float*)d_in[20];
  const float* ln_t1_b  = (const float*)d_in[21];
  const float* ln_t2_w  = (const float*)d_in[22];
  const float* ln_t2_b  = (const float*)d_in[23];
  const float* ln_f_w   = (const float*)d_in[24];
  const float* ln_f_b   = (const float*)d_in[25];
  float* out = (float*)d_out;

  // ---- workspace bump allocator (persistent region) ----
  char* base = (char*)d_ws;
  size_t off = 0;
  auto alloc = [&](size_t bytes) -> void* {
    void* r = base + off;
    off += (bytes + 255) & ~(size_t)255;
    return r;
  };

  half_t* xt16   = (half_t*)alloc((size_t)M_ * H_ * 2);
  half_t* xn16   = (half_t*)alloc((size_t)M_ * H_ * 2);
  half_t* wl16   = (half_t*)alloc((size_t)512 * 128 * 2);
  half_t* wr16   = (half_t*)alloc((size_t)512 * 128 * 2);
  half_t* win16  = (half_t*)alloc((size_t)384 * 128 * 2);
  half_t* wout16 = (half_t*)alloc((size_t)128 * 128 * 2);
  half_t* w1_16  = (half_t*)alloc((size_t)512 * 128 * 2);
  half_t* w2_16  = (half_t*)alloc((size_t)128 * 512 * 2);
  half_t* wg16   = (half_t*)alloc((size_t)128 * 256 * 2);
  float*  x_sp   = (float*) alloc((size_t)M_ * H_ * 4);
  float*  xn1    = (float*) alloc((size_t)M_ * H_ * 4);
  half_t* xn1_16 = (half_t*)alloc((size_t)M_ * H_ * 2);
  float*  x_tmp  = (float*) alloc((size_t)M_ * H_ * 4);
  half_t* cat16  = (half_t*)alloc((size_t)M_ * 256 * 2);
  float*  a_pre  = (float*) alloc((size_t)M_ * H_ * 4);

  // ---- scratch region, reused between spatial and temporal phases ----
  char* scratch = base + off;
  // spatial phase
  float* xl   = (float*)scratch;
  float* xr   = xl + (size_t)M_ * 512;
  float* wbuf = xr + (size_t)M_ * 512;
  float* den  = wbuf + (size_t)ETOT_ * BT_ * HEADS_;
  float* accS = den + (size_t)N_ * BT_ * HEADS_;
  // temporal phase (same region; spatial buffers dead by then)
  float*  qkvb     = (float*)scratch;
  float*  attn_o   = qkvb + (size_t)M_ * 384;
  float*  oproj    = attn_o + (size_t)M_ * H_;
  half_t* attn_o16 = (half_t*)(oproj + (size_t)M_ * H_);
  // FFN phase (qkv/attn dead)
  float*  f1    = (float*)scratch;
  half_t* f1_16 = (half_t*)(f1 + (size_t)M_ * 512);
  float*  f2    = (float*)((char*)f1_16 + (size_t)M_ * 512 * 2);

  const int MH = M_ * H_;

  // ---- input packing + weight conversion ----
  k_build_inputs<<<(MH + 255) / 256, 256, 0, stream>>>(x, xt16, xn16);
  k_f32_to_f16<<<256, 256, 0, stream>>>(gat_wl, wl16,   512 * 128);
  k_f32_to_f16<<<256, 256, 0, stream>>>(gat_wr, wr16,   512 * 128);
  k_f32_to_f16<<<192, 256, 0, stream>>>(ain_w,  win16,  384 * 128);
  k_f32_to_f16<<<64,  256, 0, stream>>>(aout_w, wout16, 128 * 128);
  k_f32_to_f16<<<256, 256, 0, stream>>>(w1,     w1_16,  512 * 128);
  k_f32_to_f16<<<256, 256, 0, stream>>>(w2,     w2_16,  128 * 512);
  k_f32_to_f16<<<128, 256, 0, stream>>>(gw,     wg16,   128 * 256);
  k_zero<<<64, 256, 0, stream>>>(den,  (long long)N_ * BT_ * HEADS_);
  k_zero<<<512, 256, 0, stream>>>(accS, (long long)N_ * BT_ * H_);

  // ---- spatial branch (GAT) ----
  k_gemm_wmma<0><<<gemm_grid(M_, 512), 256, 0, stream>>>(xt16, wl16, gat_bl, xl, nullptr, M_, 512, 128);
  k_gemm_wmma<0><<<gemm_grid(M_, 512), 256, 0, stream>>>(xt16, wr16, gat_br, xr, nullptr, M_, 512, 128);
  k_edge_score<<<(ETOT_ * BT_ + 7) / 8, 256, 0, stream>>>(xl, xr, ei, gat_att, wbuf, den);
  k_edge_aggr<<<ETOT_ * BT_, 128, 0, stream>>>(xl, wbuf, den, ei, accS);
  k_spatial_ln<<<M_, 128, 0, stream>>>(accS, gat_bias, ln_sp_w, ln_sp_b, x_sp);

  // ---- temporal branch (attention + FFN) ----
  k_gemm_wmma<0><<<gemm_grid(M_, 384), 256, 0, stream>>>(xn16, win16, ain_b, qkvb, nullptr, M_, 384, 128);
  k_attn<<<B_ * N_ * HEADS_, 384, 0, stream>>>(qkvb, attn_o);
  k_f32_to_f16<<<1024, 256, 0, stream>>>(attn_o, attn_o16, (long long)MH);
  k_gemm_wmma<0><<<gemm_grid(M_, 128), 256, 0, stream>>>(attn_o16, wout16, aout_b, oproj, nullptr, M_, 128, 128);
  k_add_ln<<<M_, 128, 0, stream>>>(x, oproj, ln_t1_w, ln_t1_b, xn1, xn1_16);
  k_gemm_wmma<1><<<gemm_grid(M_, 512), 256, 0, stream>>>(xn1_16, w1_16, b1, f1, f1_16, M_, 512, 128); // GELU fused
  k_gemm_wmma<0><<<gemm_grid(M_, 128), 256, 0, stream>>>(f1_16, w2_16, b2, f2, nullptr, M_, 128, 512);
  k_add_ln<<<M_, 128, 0, stream>>>(xn1, f2, ln_t2_w, ln_t2_b, x_tmp, nullptr);

  // ---- gate + final LN ----
  k_concat16<<<(M_ * 256 + 255) / 256, 256, 0, stream>>>(x_sp, x_tmp, cat16);
  k_gemm_wmma<0><<<gemm_grid(M_, 128), 256, 0, stream>>>(cat16, wg16, gb, a_pre, nullptr, M_, 128, 256);
  k_gate_final<<<M_, 128, 0, stream>>>(a_pre, x_sp, x_tmp, x, ln_f_w, ln_f_b, out);
}
